// CrossAttention_16320875725359
// MI455X (gfx1250) — compile-verified, matrix operand order checked by reading, not run on hardware
//
#include <hip/hip_runtime.h>
#include <hip/hip_bf16.h>

typedef __attribute__((ext_vector_type(16))) __bf16 v16bf;
typedef __attribute__((ext_vector_type(8)))  float  v8f;
typedef int i32x4 __attribute__((vector_size(16)));

#define WMMA_BF16(A, B, C) \
  __builtin_amdgcn_wmma_f32_16x16x32_bf16(false, (A), false, (B), (short)0, (C), false, false)

// ---------------------------------------------------------------------------
// CDNA5 async global->LDS copy (ASYNCcnt-tracked), with a guaranteed-compile
// synchronous fallback if the toolchain lacks the builtin.
// Probe-derived prototype: (global i32x4* src, local i32x4* dst, imm, imm cpol)
// ---------------------------------------------------------------------------
#if defined(__has_builtin) && __has_builtin(__builtin_amdgcn_global_load_async_to_lds_b128)
#define HAVE_ASYNC_LDS 1
#else
#define HAVE_ASYNC_LDS 0
#endif

__device__ __forceinline__ void copy16B_to_lds(__bf16* dst, const __bf16* src) {
#if HAVE_ASYNC_LDS
  __builtin_amdgcn_global_load_async_to_lds_b128(
      (__attribute__((address_space(1))) i32x4*)(__attribute__((address_space(1))) void*)(src),
      (__attribute__((address_space(3))) i32x4*)(__attribute__((address_space(3))) void*)(dst),
      0, 0);
#else
  *(uint4*)dst = *(const uint4*)src;
#endif
}

__device__ __forceinline__ void wait_async_copies() {
#if HAVE_ASYNC_LDS
#if defined(__has_builtin) && __has_builtin(__builtin_amdgcn_s_wait_asynccnt)
  __builtin_amdgcn_s_wait_asynccnt(0);
#else
  asm volatile("s_wait_asynccnt 0x0" ::: "memory");
#endif
#endif
}

// ---------------------------------------------------------------------------
// Fragment loader: 16x32 bf16 tile in "row-major pair" layout (ISA 7.12.2).
// lane = m + 16*((k/8)&1); vgpr v<4 -> k=2v+8g, v>=4 -> k=16+2(v-4)+8g.
// Works for A-fragments (row-major A) and B-fragments when the source is
// stored n-major (B^T row-major): weights (NxK), K (key x d), V^T (d x key).
// ---------------------------------------------------------------------------
template <typename PT>
__device__ __forceinline__ v16bf frag16x32(PT p, int ld, int lane) {
  const int r = lane & 15;
  const int g = (lane >> 4) << 3;       // 0 or 8
  union { uint4 u[2]; v16bf v; } U;
  U.u[0] = *(const uint4*)(p + (size_t)r * ld + g);
  U.u[1] = *(const uint4*)(p + (size_t)r * ld + g + 16);
  return U.v;
}

__device__ __forceinline__ float rmax16(float v) {
  #pragma unroll
  for (int m = 1; m < 16; m <<= 1) v = fmaxf(v, __shfl_xor(v, m, 32));
  return v;
}
__device__ __forceinline__ float rsum16(float v) {
  #pragma unroll
  for (int m = 1; m < 16; m <<= 1) v += __shfl_xor(v, m, 32);
  return v;
}

// ---------------------------------------------------------------------------
// Elementwise f32 -> bf16
// ---------------------------------------------------------------------------
__global__ void cvt_bf16_kernel(const float* __restrict__ in,
                                __bf16* __restrict__ out, int n) {
  int i = blockIdx.x * blockDim.x + threadIdx.x;
  if (i < n) out[i] = (__bf16)in[i];
}

// W (K x N) f32 -> Wt (N x K) bf16
__global__ void tconv_kernel(const float* __restrict__ W,
                             __bf16* __restrict__ Wt, int K, int N) {
  int n = blockIdx.x * 16 + threadIdx.x;
  int k = blockIdx.y * 16 + threadIdx.y;
  if (n < N && k < K) Wt[(size_t)n * K + k] = (__bf16)W[(size_t)k * N + n];
}

// ---------------------------------------------------------------------------
// GEMM: C = A(MxK) * Bt(NxK)^T + bias, bf16 in, f32 accumulate.
// 128x128 tile per block, BK=32, 256 threads = 8 waves; wave tile 32x64
// (2 A-frags x 4 B-frags = 8 WMMAs per K-step). Double-buffered LDS with
// async global->LDS copies overlapping WMMA compute.
// MODE 0: bf16 row-major. MODE 1: f32 row-major (final). MODE 2: per-head
// transposed Vt (B,H,128,512) bf16.
// ---------------------------------------------------------------------------
template <int MODE>
__global__ __launch_bounds__(256)
void gemm_bf16_kernel(const __bf16* __restrict__ A,
                      const __bf16* __restrict__ Bt,
                      const float* __restrict__ bias,
                      void* __restrict__ Cout, int M, int N, int K) {
  __shared__ __align__(16) __bf16 As[2][128 * 32];
  __shared__ __align__(16) __bf16 Bs[2][128 * 32];

  const int tid  = threadIdx.x;
  const int lane = tid & 31;
  const int wid  = tid >> 5;
  const int m0 = blockIdx.y * 128, n0 = blockIdx.x * 128;
  const int wm = (wid >> 1) * 32;      // 0,32,64,96
  const int wn = (wid & 1) * 64;       // 0,64

  v8f acc[2][4] = {};

  const int srow = tid >> 1;           // 0..127
  const int scol = (tid & 1) * 16;     // 0 or 16

  auto load_tiles = [&](int buf, int kt) {
    const __bf16* asrc = A  + (size_t)(m0 + srow) * K + kt + scol;
    const __bf16* bsrc = Bt + (size_t)(n0 + srow) * K + kt + scol;
    copy16B_to_lds(&As[buf][srow * 32 + scol],     asrc);
    copy16B_to_lds(&As[buf][srow * 32 + scol + 8], asrc + 8);
    copy16B_to_lds(&Bs[buf][srow * 32 + scol],     bsrc);
    copy16B_to_lds(&Bs[buf][srow * 32 + scol + 8], bsrc + 8);
  };

  load_tiles(0, 0);
  wait_async_copies();
  __syncthreads();

  int buf = 0;
  for (int kt = 0; kt < K; kt += 32) {
    if (kt + 32 < K) load_tiles(buf ^ 1, kt + 32);   // prefetch next tile

    v16bf a0 = frag16x32(&As[buf][(wm)      * 32], 32, lane);
    v16bf a1 = frag16x32(&As[buf][(wm + 16) * 32], 32, lane);
    v16bf b[4];
    #pragma unroll
    for (int j = 0; j < 4; ++j)
      b[j] = frag16x32(&Bs[buf][(wn + j * 16) * 32], 32, lane);

    #pragma unroll
    for (int j = 0; j < 4; ++j) {
      acc[0][j] = WMMA_BF16(a0, b[j], acc[0][j]);
      acc[1][j] = WMMA_BF16(a1, b[j], acc[1][j]);
    }

    wait_async_copies();
    __syncthreads();
    buf ^= 1;
  }

  // Epilogue. C-frag layout: lane = n + 16*(m/8), vgpr = m%8.
  #pragma unroll
  for (int i = 0; i < 2; ++i) {
    #pragma unroll
    for (int j = 0; j < 4; ++j) {
      const int col = n0 + wn + j * 16 + (lane & 15);
      const float bv = bias[col];
      #pragma unroll
      for (int v = 0; v < 8; ++v) {
        const int row = m0 + wm + i * 16 + v + 8 * (lane >> 4);
        const float val = acc[i][j][v] + bv;
        if (MODE == 0) {
          ((__bf16*)Cout)[(size_t)row * N + col] = (__bf16)val;
        } else if (MODE == 1) {
          ((float*)Cout)[(size_t)row * N + col] = val;
        } else {
          // row = b*512 + sy ; col = h*128 + d  ->  Vt[b][h][d][sy]
          const int bb = row >> 9, sy = row & 511;
          const int hh = col >> 7, dd = col & 127;
          ((__bf16*)Cout)[(((size_t)(bb * 8 + hh) * 128 + dd) << 9) + sy] = (__bf16)val;
        }
      }
    }
  }
}

// ---------------------------------------------------------------------------
// Flash attention: one wave per (b, h, 16-query block). Sy=512, Dh=128.
// Q frags in registers; K B-frags straight from global; online softmax;
// P converted C-frag -> A-frag via a 1 KB LDS bounce; P*Vt via 8 WMMAs/chunk.
// ---------------------------------------------------------------------------
__global__ __launch_bounds__(32)
void attn_kernel(const __bf16* __restrict__ Q,
                 const __bf16* __restrict__ K,
                 const __bf16* __restrict__ Vt,
                 __bf16* __restrict__ O) {
  const int lane = threadIdx.x;
  const int blk  = blockIdx.x;          // b*2048 + h*256 + qb
  const int qb   = blk & 255;
  const int h    = (blk >> 8) & 7;
  const int b    = blk >> 11;

  const __bf16* Qbase = Q  + ((size_t)(b * 4096 + qb * 16)) * 1024 + h * 128;
  const __bf16* Kbase = K  + ((size_t)(b * 512)) * 1024 + h * 128;
  const __bf16* Vbase = Vt + ((size_t)(b * 8 + h)) * 128 * 512;

  v16bf qf[4];
  #pragma unroll
  for (int kc = 0; kc < 4; ++kc) qf[kc] = frag16x32(Qbase + kc * 32, 1024, lane);

  v8f o[8] = {};
  float mi[8], li[8];
  #pragma unroll
  for (int v = 0; v < 8; ++v) { mi[v] = -1e30f; li[v] = 0.f; }

  __shared__ __align__(16) __bf16 P[16 * 32];
  const float scale = 0.08838834764831845f;   // 1/sqrt(128)
  const int g = lane >> 4, nl = lane & 15;

  for (int j = 0; j < 16; ++j) {
    const int key0 = j * 32;
    v8f s0 = {}, s1 = {};
    #pragma unroll
    for (int kc = 0; kc < 4; ++kc) {
      v16bf kb0 = frag16x32(Kbase + (size_t)(key0)      * 1024 + kc * 32, 1024, lane);
      v16bf kb1 = frag16x32(Kbase + (size_t)(key0 + 16) * 1024 + kc * 32, 1024, lane);
      s0 = WMMA_BF16(qf[kc], kb0, s0);
      s1 = WMMA_BF16(qf[kc], kb1, s1);
    }

    // Online softmax update; lane owns rows m = v + 8*g of the 16-query block.
    #pragma unroll
    for (int v = 0; v < 8; ++v) {
      const float sv0 = s0[v] * scale;
      const float sv1 = s1[v] * scale;
      const float rmax = rmax16(fmaxf(sv0, sv1));
      const float mn   = fmaxf(mi[v], rmax);
      const float alpha = __expf(mi[v] - mn);
      const float p0 = __expf(sv0 - mn);
      const float p1 = __expf(sv1 - mn);
      li[v] = li[v] * alpha + rsum16(p0 + p1);
      mi[v] = mn;
      #pragma unroll
      for (int ds = 0; ds < 8; ++ds) o[ds][v] *= alpha;
      const int m = v + 8 * g;
      P[m * 32 + nl]      = (__bf16)p0;
      P[m * 32 + 16 + nl] = (__bf16)p1;
    }

    // Same-wave LDS stores -> loads: drain DScnt and fence the compiler.
    asm volatile("s_wait_dscnt 0x0" ::: "memory");

    v16bf pf = frag16x32(&P[0], 32, lane);          // 16x32 A-frag of probs
    #pragma unroll
    for (int ds = 0; ds < 8; ++ds) {
      v16bf vf = frag16x32(Vbase + (size_t)(ds * 16) * 512 + key0, 512, lane);
      o[ds] = WMMA_BF16(pf, vf, o[ds]);
    }
  }

  __bf16* Obase = O + ((size_t)(b * 4096 + qb * 16)) * 1024 + h * 128;
  #pragma unroll
  for (int v = 0; v < 8; ++v) {
    const int m = v + 8 * g;
    const float inv = 1.f / li[v];
    #pragma unroll
    for (int ds = 0; ds < 8; ++ds) {
      const int d = ds * 16 + nl;
      Obase[(size_t)m * 1024 + d] = (__bf16)(o[ds][v] * inv);
    }
  }
}

// ---------------------------------------------------------------------------
// Host side
// ---------------------------------------------------------------------------
extern "C" void kernel_launch(void* const* d_in, const int* in_sizes, int n_in,
                              void* d_out, int out_size, void* d_ws, size_t ws_size,
                              hipStream_t stream) {
  (void)in_sizes; (void)n_in; (void)out_size; (void)ws_size;

  const float* x  = (const float*)d_in[0];   // (4,4096,1024)
  const float* y  = (const float*)d_in[1];   // (4,512,1,768) -> (2048,768)
  const float* Wq = (const float*)d_in[2];
  const float* bq = (const float*)d_in[3];
  const float* Wk = (const float*)d_in[4];
  const float* bk = (const float*)d_in[5];
  const float* Wv = (const float*)d_in[6];
  const float* bv = (const float*)d_in[7];
  const float* Wo = (const float*)d_in[8];
  const float* bo = (const float*)d_in[9];
  float* out = (float*)d_out;

  size_t off = 0;
  auto alloc = [&](size_t bytes) -> void* {
    void* p = (char*)d_ws + off;
    off += (bytes + 255) & ~(size_t)255;
    return p;
  };
  __bf16* xbf = (__bf16*)alloc((size_t)16384 * 1024 * 2);
  __bf16* ybf = (__bf16*)alloc((size_t)2048 * 768 * 2);
  __bf16* Wqt = (__bf16*)alloc((size_t)1024 * 1024 * 2);
  __bf16* Wkt = (__bf16*)alloc((size_t)1024 * 768 * 2);
  __bf16* Wvt = (__bf16*)alloc((size_t)1024 * 768 * 2);
  __bf16* Wot = (__bf16*)alloc((size_t)1024 * 1024 * 2);
  __bf16* Qb  = (__bf16*)alloc((size_t)16384 * 1024 * 2);
  __bf16* Kb  = (__bf16*)alloc((size_t)2048 * 1024 * 2);
  __bf16* Vtb = (__bf16*)alloc((size_t)4 * 8 * 128 * 512 * 2);
  __bf16* A2  = (__bf16*)alloc((size_t)16384 * 1024 * 2);

  // 1) Convert activations to bf16
  {
    int n = 16384 * 1024;
    cvt_bf16_kernel<<<(n + 255) / 256, 256, 0, stream>>>(x, xbf, n);
    n = 2048 * 768;
    cvt_bf16_kernel<<<(n + 255) / 256, 256, 0, stream>>>(y, ybf, n);
  }
  // 2) Transpose-convert weights to (N x K) bf16
  {
    dim3 blk(16, 16);
    tconv_kernel<<<dim3(64, 64), blk, 0, stream>>>(Wq, Wqt, 1024, 1024);
    tconv_kernel<<<dim3(64, 48), blk, 0, stream>>>(Wk, Wkt, 768, 1024);
    tconv_kernel<<<dim3(64, 48), blk, 0, stream>>>(Wv, Wvt, 768, 1024);
    tconv_kernel<<<dim3(64, 64), blk, 0, stream>>>(Wo, Wot, 1024, 1024);
  }
  // 3) Projections (128x128 tiles)
  gemm_bf16_kernel<0><<<dim3(8, 128), 256, 0, stream>>>(xbf, Wqt, bq, Qb, 16384, 1024, 1024);
  gemm_bf16_kernel<0><<<dim3(8, 16),  256, 0, stream>>>(ybf, Wkt, bk, Kb, 2048, 1024, 768);
  gemm_bf16_kernel<2><<<dim3(8, 16),  256, 0, stream>>>(ybf, Wvt, bv, Vtb, 2048, 1024, 768);
  // 4) Attention: B*H*(Sx/16) = 8192 one-wave blocks
  attn_kernel<<<8192, 32, 0, stream>>>(Qb, Kb, Vtb, A2);
  // 5) Output projection (f32 out)
  gemm_bf16_kernel<1><<<dim3(8, 128), 256, 0, stream>>>(A2, Wot, bo, out, 16384, 1024, 1024);
}
